// MultiHeadAttention_72069551227273
// MI455X (gfx1250) — matrix operand order repaired and checked by reading, not compiled, not run on hardware
//
#include <hip/hip_runtime.h>
#include <cstdint>

// ---------------------------------------------------------------------------
// Types / helpers
// ---------------------------------------------------------------------------
typedef __attribute__((ext_vector_type(16))) __bf16          bf16x16;
typedef __attribute__((ext_vector_type(8)))  float           f32x8;
typedef __attribute__((ext_vector_type(8)))  unsigned short  u16x8;

union AB {            // one WMMA A/B fragment (16 bf16 halves = 8 VGPRs)
  bf16x16 v;
  u16x8   h[2];
};

// gfx1250 async global->LDS path (ASYNCcnt), with portable fallback
#if defined(__has_builtin)
#  if __has_builtin(__builtin_amdgcn_global_load_async_to_lds_b128)
#    define HAVE_ASYNC_LDS 1
#  endif
#endif
#ifndef HAVE_ASYNC_LDS
#  define HAVE_ASYNC_LDS 0
#endif

#if HAVE_ASYNC_LDS
typedef int i32x4 __attribute__((vector_size(4 * sizeof(int))));
typedef __attribute__((address_space(1))) i32x4 as1_i32x4;   // global
typedef __attribute__((address_space(3))) i32x4 as3_i32x4;   // LDS
__device__ __forceinline__ void async_copy16(const unsigned short* g,
                                             unsigned short* l) {
  __builtin_amdgcn_global_load_async_to_lds_b128(
      (as1_i32x4*)(uintptr_t)g, (as3_i32x4*)(uintptr_t)l, 0, 0);
}
__device__ __forceinline__ void async_wait0() {
  asm volatile("s_wait_asynccnt 0x0" ::: "memory");
}
#endif

__device__ __forceinline__ f32x8 zero8() {
  f32x8 z = {0.f, 0.f, 0.f, 0.f, 0.f, 0.f, 0.f, 0.f};
  return z;
}

__device__ __forceinline__ unsigned short f2bf(float f) {
  unsigned int u = __float_as_uint(f);
  u += 0x7FFFu + ((u >> 16) & 1u);          // round-to-nearest-even
  return (unsigned short)(u >> 16);
}

// ---------------------------------------------------------------------------
// Precision conversion kernels
// ---------------------------------------------------------------------------
__global__ void k_f32_to_bf16(const float* __restrict__ x,
                              unsigned short* __restrict__ y, int n) {
  int i = blockIdx.x * blockDim.x + threadIdx.x;
  if (i < n) y[i] = f2bf(x[i]);
}

// W [R][C] f32 (row-major) -> Wt [C][R] bf16  (R, C multiples of 32)
__global__ __launch_bounds__(256) void k_transpose_bf16(
    const float* __restrict__ W, unsigned short* __restrict__ Wt, int R, int C) {
  __shared__ float tile[32][33];
  int c0 = blockIdx.x * 32, r0 = blockIdx.y * 32;
  int tx = threadIdx.x & 31, ty = threadIdx.x >> 5;   // 256 thr: ty = 0..7
#pragma unroll
  for (int i = 0; i < 32; i += 8)
    tile[ty + i][tx] = W[(size_t)(r0 + ty + i) * C + c0 + tx];
  __syncthreads();
#pragma unroll
  for (int i = 0; i < 32; i += 8)
    Wt[(size_t)(c0 + ty + i) * R + r0 + tx] = f2bf(tile[tx][ty + i]);
}

// bf[n] = sum_j b1[j] * W[j][n] + b2[n]   (W f32 [J][N] row-major)
__global__ void k_fuse_bias(const float* __restrict__ b1, const float* __restrict__ W,
                            const float* __restrict__ b2, float* __restrict__ bf,
                            int J, int N) {
  int n = blockIdx.x * blockDim.x + threadIdx.x;
  if (n >= N) return;
  float s = b2[n];
  for (int j = 0; j < J; ++j) s += b1[j] * W[(size_t)j * N + n];
  bf[n] = s;
}

// ---------------------------------------------------------------------------
// Generic bf16 WMMA GEMM:  C[m][n] = (sum_k A[m][k]*Bt[n][k] + bias[n]) * scale
// A  : [M][K] bf16 row-major          (M mult of 128, K mult of 64)
// Bt : [N][K] bf16 row-major (B^T)    (N mult of 64)
// Workgroup = 8 waves; tile = 128(M) x 64(N); Bt staged K=64 per barrier pair
// (async global->LDS when available), LDS stride padded to 72 (no bank hits).
// ---------------------------------------------------------------------------
template <bool OUT_BF16>
__global__ __launch_bounds__(256) void k_gemm_bf16(
    const unsigned short* __restrict__ A, const unsigned short* __restrict__ Bt,
    const float* __restrict__ bias, void* __restrict__ Cout,
    int M, int N, int K, float scale) {
  __shared__ unsigned short bl[64 * 72];          // 64 N-rows x 64 K, stride 72
  const int lane = threadIdx.x & 31;
  const int w    = threadIdx.x >> 5;
  const int nl   = lane & 15, kh = lane >> 4;
  const int nbase = blockIdx.x * 64;
  const int mbase = blockIdx.y * 128;
  const int mrow  = mbase + w * 16 + nl;          // A-fragment row for this lane

  f32x8 acc[4];
#pragma unroll
  for (int t = 0; t < 4; ++t) acc[t] = zero8();

  for (int k0 = 0; k0 < K; k0 += 64) {
    __syncthreads();
    // stage Bt[nbase..+63][k0..+63]: 512 x 16B chunks, 2 per thread
#pragma unroll
    for (int s = 0; s < 2; ++s) {
      int slot = threadIdx.x + s * 256;
      int row = slot >> 3, ch = (slot & 7) * 8;
      const unsigned short* src = &Bt[(size_t)(nbase + row) * K + k0 + ch];
      unsigned short* dst = &bl[row * 72 + ch];
#if HAVE_ASYNC_LDS
      async_copy16(src, dst);
#else
      *(u16x8*)dst = *(const u16x8*)src;
#endif
    }
#if HAVE_ASYNC_LDS
    async_wait0();
#endif
    __syncthreads();

#pragma unroll
    for (int ks = 0; ks < 2; ++ks) {
      AB a;   // A layout: half-wave kh -> K {0..7,16..23} / {8..15,24..31}
      a.h[0] = *(const u16x8*)&A[(size_t)mrow * K + k0 + ks * 32 + 8 * kh];
      a.h[1] = *(const u16x8*)&A[(size_t)mrow * K + k0 + ks * 32 + 16 + 8 * kh];
#pragma unroll
      for (int t = 0; t < 4; ++t) {
        AB b; // B layout: lane = column, half-wave kh -> K {0..7,16..23} / {8..15,24..31}
        const unsigned short* r = &bl[(t * 16 + nl) * 72 + ks * 32 + 8 * kh];
        b.h[0] = *(const u16x8*)&r[0];
        b.h[1] = *(const u16x8*)&r[16];
        acc[t] = __builtin_amdgcn_wmma_f32_16x16x32_bf16(
            false, a.v, false, b.v, (short)0, acc[t], false, false);
      }
    }
  }

#pragma unroll
  for (int t = 0; t < 4; ++t) {
    int col  = nbase + t * 16 + nl;
    float bv = bias ? bias[col] : 0.0f;
#pragma unroll
    for (int r = 0; r < 8; ++r) {
      int row = mbase + w * 16 + r + 8 * kh;      // C/D layout
      float val = (acc[t][r] + bv) * scale;
      if (OUT_BF16)
        ((unsigned short*)Cout)[(size_t)row * N + col] = f2bf(val);
      else
        ((float*)Cout)[(size_t)row * N + col] = val;
    }
  }
}

// ---------------------------------------------------------------------------
// Flash attention (causal).  Qh/Kh/Vh: bf16 [B*S][1024], head h = cols h*64..
// Q is pre-scaled by 1/sqrt(d_k).  One workgroup = 128 query rows of one (b,h),
// 8 waves x 16-row strips; 64-key blocks staged in LDS (K via async path);
// online softmax; P round-trips through per-wave LDS to re-layout C/D -> A.
// ---------------------------------------------------------------------------
__global__ __launch_bounds__(256) void k_flash_attn(
    const unsigned short* __restrict__ Qh, const unsigned short* __restrict__ Kh,
    const unsigned short* __restrict__ Vh, unsigned short* __restrict__ Oc,
    int S, int Demb) {
  const int qblk = blockIdx.x * 128;
  const int h    = blockIdx.y;
  const int b    = blockIdx.z;
  const int lane = threadIdx.x & 31, w = threadIdx.x >> 5;
  const int nl = lane & 15, kh = lane >> 4;

  __shared__ unsigned short Kl[64 * 72];       // [key][dim]   stride 72
  __shared__ unsigned short Vt[64 * 72];       // [dim][key]   stride 72
  __shared__ unsigned short Pl[8 * 16 * 72];   // per-wave P   [qrow][key]

  const size_t ld   = (size_t)Demb;
  const size_t base = ((size_t)b * S) * ld + (size_t)h * 64;

  // Q fragments (d_k = 64 -> two K=32 steps), A layout
  AB qf[2];
  {
    const unsigned short* qr = Qh + base + (size_t)(qblk + w * 16 + nl) * ld;
    qf[0].h[0] = *(const u16x8*)&qr[0 + 8 * kh];
    qf[0].h[1] = *(const u16x8*)&qr[16 + 8 * kh];
    qf[1].h[0] = *(const u16x8*)&qr[32 + 8 * kh];
    qf[1].h[1] = *(const u16x8*)&qr[48 + 8 * kh];
  }

  f32x8 oacc[4];
#pragma unroll
  for (int t = 0; t < 4; ++t) oacc[t] = zero8();
  float rm[8], rl[8];
#pragma unroll
  for (int r = 0; r < 8; ++r) { rm[r] = -1e30f; rl[r] = 0.0f; }

  const int trow = threadIdx.x >> 2;           // staging: 4 threads per key row
  const int tch  = (threadIdx.x & 3) * 8;
  const int nkb  = (qblk + 128) / 64;          // causal: keys <= qblk+127

  for (int kb = 0; kb < nkb; ++kb) {
    const int kblk = kb * 64;
    __syncthreads();
    {
      const size_t gro = base + (size_t)(kblk + trow) * ld + tch;
#if HAVE_ASYNC_LDS
      async_copy16(&Kh[gro], &Kl[trow * 72 + tch]);   // K: direct async DMA
#else
      *(u16x8*)&Kl[trow * 72 + tch] = *(const u16x8*)&Kh[gro];
#endif
      u16x8 vv = *(const u16x8*)&Vh[gro];             // V: transpose into LDS
#pragma unroll
      for (int i = 0; i < 8; ++i) Vt[(tch + i) * 72 + trow] = vv[i];
#if HAVE_ASYNC_LDS
      async_wait0();
#endif
    }
    __syncthreads();

    // ---- scores S = Q K^T  (already scaled via Q) -------------------------
    f32x8 sacc[4];
#pragma unroll
    for (int t = 0; t < 4; ++t) {
      sacc[t] = zero8();
#pragma unroll
      for (int ks = 0; ks < 2; ++ks) {
        AB bf;
        const unsigned short* r = &Kl[(t * 16 + nl) * 72 + ks * 32 + 8 * kh];
        bf.h[0] = *(const u16x8*)&r[0];
        bf.h[1] = *(const u16x8*)&r[16];
        sacc[t] = __builtin_amdgcn_wmma_f32_16x16x32_bf16(
            false, qf[ks].v, false, bf.v, (short)0, sacc[t], false, false);
      }
    }

    // ---- causal mask + block row-max --------------------------------------
    float bm[8];
#pragma unroll
    for (int r = 0; r < 8; ++r) bm[r] = -1e30f;
    const int qrow0 = qblk + w * 16 + 8 * kh;
#pragma unroll
    for (int t = 0; t < 4; ++t) {
      int col = kblk + t * 16 + nl;
#pragma unroll
      for (int r = 0; r < 8; ++r) {
        float s = (col <= qrow0 + r) ? sacc[t][r] : -1e30f;
        sacc[t][r] = s;
        bm[r] = fmaxf(bm[r], s);
      }
    }
#pragma unroll
    for (int r = 0; r < 8; ++r) {     // row lives in one 16-lane half
      float m = bm[r];
      m = fmaxf(m, __shfl_xor(m, 1, 32));
      m = fmaxf(m, __shfl_xor(m, 2, 32));
      m = fmaxf(m, __shfl_xor(m, 4, 32));
      m = fmaxf(m, __shfl_xor(m, 8, 32));
      bm[r] = m;
    }
    float corr[8];
#pragma unroll
    for (int r = 0; r < 8; ++r) {
      float nm = fmaxf(rm[r], bm[r]);
      corr[r]  = __expf(rm[r] - nm);
      rm[r]    = nm;
    }

    // ---- P = exp(S - m), row sums, spill P to LDS in A-readable layout ----
    float bs[8];
#pragma unroll
    for (int r = 0; r < 8; ++r) bs[r] = 0.0f;
    unsigned short* pw = &Pl[w * 16 * 72];
#pragma unroll
    for (int t = 0; t < 4; ++t) {
#pragma unroll
      for (int r = 0; r < 8; ++r) {
        float pv = __expf(sacc[t][r] - rm[r]);
        bs[r] += pv;
        pw[(r + 8 * kh) * 72 + t * 16 + nl] = f2bf(pv);
      }
    }
#pragma unroll
    for (int r = 0; r < 8; ++r) {
      float s = bs[r];
      s += __shfl_xor(s, 1, 32);
      s += __shfl_xor(s, 2, 32);
      s += __shfl_xor(s, 4, 32);
      s += __shfl_xor(s, 8, 32);
      rl[r] = rl[r] * corr[r] + s;
    }
#pragma unroll
    for (int t = 0; t < 4; ++t)
#pragma unroll
      for (int r = 0; r < 8; ++r) oacc[t][r] *= corr[r];

    // ---- O += P @ V  (A = P from LDS, B = V^T from LDS) -------------------
#pragma unroll
    for (int ks = 0; ks < 2; ++ks) {
      AB pa;
      const unsigned short* pr = &pw[nl * 72 + ks * 32];
      pa.h[0] = *(const u16x8*)&pr[8 * kh];
      pa.h[1] = *(const u16x8*)&pr[16 + 8 * kh];
#pragma unroll
      for (int t = 0; t < 4; ++t) {
        AB vb;
        const unsigned short* vr = &Vt[(t * 16 + nl) * 72 + ks * 32 + 8 * kh];
        vb.h[0] = *(const u16x8*)&vr[0];
        vb.h[1] = *(const u16x8*)&vr[16];
        oacc[t] = __builtin_amdgcn_wmma_f32_16x16x32_bf16(
            false, pa.v, false, vb.v, (short)0, oacc[t], false, false);
      }
    }
  }

  // ---- normalize + store concat head slice (bf16) -------------------------
#pragma unroll
  for (int r = 0; r < 8; ++r) rl[r] = 1.0f / rl[r];
#pragma unroll
  for (int t = 0; t < 4; ++t) {
#pragma unroll
    for (int r = 0; r < 8; ++r) {
      int qrow = qblk + w * 16 + r + 8 * kh;
      Oc[base + (size_t)qrow * ld + t * 16 + nl] = f2bf(oacc[t][r] * rl[r]);
    }
  }
}

// ---------------------------------------------------------------------------
// Launch
// ---------------------------------------------------------------------------
extern "C" void kernel_launch(void* const* d_in, const int* in_sizes, int n_in,
                              void* d_out, int out_size, void* d_ws, size_t ws_size,
                              hipStream_t stream) {
  (void)in_sizes; (void)n_in; (void)out_size; (void)ws_size;
  const int B = 2, S = 2048, DM = 1024, DE = 1024, H = 16;
  const size_t M = (size_t)B * S;                     // 4096 rows

  const float* q    = (const float*)d_in[0];
  const float* k    = (const float*)d_in[1];
  const float* v    = (const float*)d_in[2];
  /* d_in[3] = mask: exactly tril(ones) -> applied analytically */
  const float* W_in = (const float*)d_in[4];
  const float* b_in = (const float*)d_in[5];
  const float* Wq   = (const float*)d_in[6];
  const float* bq   = (const float*)d_in[7];
  const float* Wk   = (const float*)d_in[8];
  const float* bk   = (const float*)d_in[9];
  const float* Wv   = (const float*)d_in[10];
  const float* bv   = (const float*)d_in[11];
  const float* Wo   = (const float*)d_in[12];
  const float* bo   = (const float*)d_in[13];

  char* p = (char*)d_ws;
  auto take = [&p](size_t bytes) -> char* {
    char* r = p; p += (bytes + 255) & ~(size_t)255; return r;
  };
  unsigned short* qb   = (unsigned short*)take(M * DM * 2);
  unsigned short* kb   = (unsigned short*)take(M * DM * 2);
  unsigned short* vb   = (unsigned short*)take(M * DM * 2);
  unsigned short* Winb = (unsigned short*)take((size_t)DM * DE * 2);
  unsigned short* WqT  = (unsigned short*)take((size_t)DE * DE * 2);
  unsigned short* WkT  = (unsigned short*)take((size_t)DE * DE * 2);
  unsigned short* WvT  = (unsigned short*)take((size_t)DE * DE * 2);
  unsigned short* WoT  = (unsigned short*)take((size_t)DE * DM * 2);
  unsigned short* WfqT = (unsigned short*)take((size_t)DE * DM * 2);
  unsigned short* WfkT = (unsigned short*)take((size_t)DE * DM * 2);
  unsigned short* WfvT = (unsigned short*)take((size_t)DE * DM * 2);
  float* bfq = (float*)take((size_t)DE * 4);
  float* bfk = (float*)take((size_t)DE * 4);
  float* bfv = (float*)take((size_t)DE * 4);
  unsigned short* Qh = (unsigned short*)take(M * DE * 2);
  unsigned short* Kh = (unsigned short*)take(M * DE * 2);
  unsigned short* Vh = (unsigned short*)take(M * DE * 2);
  unsigned short* Oc = (unsigned short*)take(M * DE * 2);

  // 1) f32 -> bf16 conversions
  {
    int n = (int)(M * DM);
    k_f32_to_bf16<<<dim3((n + 255) / 256), 256, 0, stream>>>(q, qb, n);
    k_f32_to_bf16<<<dim3((n + 255) / 256), 256, 0, stream>>>(k, kb, n);
    k_f32_to_bf16<<<dim3((n + 255) / 256), 256, 0, stream>>>(v, vb, n);
    int nw = DM * DE;
    k_f32_to_bf16<<<dim3((nw + 255) / 256), 256, 0, stream>>>(W_in, Winb, nw);
  }
  // 2) transpose-convert per-head / output weights
  k_transpose_bf16<<<dim3(DE / 32, DE / 32), 256, 0, stream>>>(Wq, WqT, DE, DE);
  k_transpose_bf16<<<dim3(DE / 32, DE / 32), 256, 0, stream>>>(Wk, WkT, DE, DE);
  k_transpose_bf16<<<dim3(DE / 32, DE / 32), 256, 0, stream>>>(Wv, WvT, DE, DE);
  k_transpose_bf16<<<dim3(DM / 32, DE / 32), 256, 0, stream>>>(Wo, WoT, DE, DM);
  // 3) fused biases  bf* = b_in @ W* + b*
  k_fuse_bias<<<dim3(DE / 256), 256, 0, stream>>>(b_in, Wq, bq, bfq, DE, DE);
  k_fuse_bias<<<dim3(DE / 256), 256, 0, stream>>>(b_in, Wk, bk, bfk, DE, DE);
  k_fuse_bias<<<dim3(DE / 256), 256, 0, stream>>>(b_in, Wv, bv, bfv, DE, DE);
  // 4) fused weights  Wf*T[n][kmod] = sum_j W*T[n][j] * W_in[kmod][j]
  k_gemm_bf16<true><<<dim3(DM / 64, DE / 128), 256, 0, stream>>>(
      WqT, Winb, nullptr, WfqT, DE, DM, DE, 1.0f);
  k_gemm_bf16<true><<<dim3(DM / 64, DE / 128), 256, 0, stream>>>(
      WkT, Winb, nullptr, WfkT, DE, DM, DE, 1.0f);
  k_gemm_bf16<true><<<dim3(DM / 64, DE / 128), 256, 0, stream>>>(
      WvT, Winb, nullptr, WfvT, DE, DM, DE, 1.0f);
  // 5) projections (Q pre-scaled by 1/sqrt(d_k) = 0.125)
  k_gemm_bf16<true><<<dim3(DE / 64, (unsigned)(M / 128)), 256, 0, stream>>>(
      qb, WfqT, bfq, Qh, (int)M, DE, DM, 0.125f);
  k_gemm_bf16<true><<<dim3(DE / 64, (unsigned)(M / 128)), 256, 0, stream>>>(
      kb, WfkT, bfk, Kh, (int)M, DE, DM, 1.0f);
  k_gemm_bf16<true><<<dim3(DE / 64, (unsigned)(M / 128)), 256, 0, stream>>>(
      vb, WfvT, bfv, Vh, (int)M, DE, DM, 1.0f);
  // 6) causal flash attention -> concat buffer
  k_flash_attn<<<dim3(S / 128, H, B), 256, 0, stream>>>(Qh, Kh, Vh, Oc, S, DE);
  // 7) output projection (f32 result)
  k_gemm_bf16<false><<<dim3(DM / 64, (unsigned)(M / 128)), 256, 0, stream>>>(
      Oc, WoT, bo, (float*)d_out, (int)M, DM, DE, 1.0f);
}
